// MandiFlowNet_85315230368283
// MI455X (gfx1250) — matrix operand order repaired and checked by reading, not compiled, not run on hardware
//
#include <hip/hip_runtime.h>
#include <hip/hip_bf16.h>
#include <math.h>

// ---------------- problem constants (match reference) ----------------
#define NNODES 50000
#define NEDGES 800000
#define FIN    128
#define HDIM   64
#define GATES  256        // 4*HDIM

typedef __attribute__((ext_vector_type(16))) _Float16     v16h;
typedef __attribute__((ext_vector_type(8)))  float        v8f;
typedef __attribute__((ext_vector_type(4)))  unsigned int u32x4;
typedef __attribute__((ext_vector_type(4)))  float        f32x4;

union frag16 { u32x4 q[2]; v16h v; };   // 32 bytes: one WMMA f16 A/B fragment per lane

static __device__ __forceinline__ v8f wmma_f16(v16h a, v16h b, v8f c) {
  // D = A(16x32 f16) * B(32x16 f16) + C(16x16 f32)
  return __builtin_amdgcn_wmma_f32_16x16x32_f16(false, a, false, b, (short)0, c,
                                                false, false);
}

// ---------------- small conversion / init kernels ----------------
__global__ void k_cvt_f16(const float* __restrict__ in, _Float16* __restrict__ out, int n) {
  int i = blockIdx.x * blockDim.x + threadIdx.x;
  if (i < n) out[i] = (_Float16)in[i];
}

__global__ void k_relu_cvt_f16(const float* __restrict__ in, _Float16* __restrict__ out, int n) {
  int i = blockIdx.x * blockDim.x + threadIdx.x;
  if (i < n) { float v = in[i]; out[i] = (_Float16)(v > 0.f ? v : 0.f); }
}

// W[K][Nc] (row-major) -> Wt[Nc][K] f16
__global__ void k_cvt_transpose_f16(const float* __restrict__ W, _Float16* __restrict__ Wt,
                                    int K, int Nc) {
  int i = blockIdx.x * blockDim.x + threadIdx.x;
  if (i >= K * Nc) return;
  int k = i / Nc, n = i - k * Nc;
  Wt[(size_t)n * K + k] = (_Float16)W[i];
}

__global__ void k_fill(float* __restrict__ p, float v, int n) {
  int i = blockIdx.x * blockDim.x + threadIdx.x;
  if (i < n) p[i] = v;
}

// agg[n][f] = bias[f]
__global__ void k_agg_init(float* __restrict__ agg, const float* __restrict__ b, int n) {
  int i = blockIdx.x * blockDim.x + threadIdx.x;
  if (i < n) agg[i] = b[i & (HDIM - 1)];
}

// ---------------- degree / normalization ----------------
__global__ void k_deg_accum(const long long* __restrict__ col, const float* __restrict__ ew,
                            float* __restrict__ deg, int e_cnt) {
  int i = blockIdx.x * blockDim.x + threadIdx.x;
  if (i < e_cnt) atomicAdd(&deg[(int)col[i]], ew[i]);
}

__global__ void k_deg_invsqrt(const float* __restrict__ deg, float* __restrict__ dis, int n) {
  int i = blockIdx.x * blockDim.x + threadIdx.x;
  if (i < n) {
    float d = deg[i];
    dis[i] = d > 0.f ? rsqrtf(fmaxf(d, 1e-12f)) : 0.f;
  }
}

// ---------------- WMMA GEMM: C[M][Nout] = A_f16[M][K] @ Bt_f16[Nout][K]^T ----------------
// block = 128 threads (4 waves); wave w handles col tile (blockIdx.y*4 + w); blockIdx.x = row tile
__global__ void k_gemm_f16(const _Float16* __restrict__ A, const _Float16* __restrict__ Bt,
                           float* __restrict__ C, int K, int Nout) {
  const int wave = threadIdx.x >> 5;
  const int lane = threadIdx.x & 31;
  const int m    = lane & 15;
  const int hh   = lane >> 4;
  const int n0   = (blockIdx.y * 4 + wave) * 16;
  const int r0   = blockIdx.x * 16;

  const _Float16* arow = A + (size_t)(r0 + m) * K;
  const _Float16* brow = Bt + (size_t)(n0 + m) * K;

  v8f acc = {};
  for (int k0 = 0; k0 < K; k0 += 32) {
    frag16 a, b;
    // A 16x32 f16 fragment: lane(m,hh): K = k0 + hh*8 + e  and  k0 + 16 + hh*8 + e
    a.q[0] = *(const u32x4*)(arow + k0 + hh * 8);
    a.q[1] = *(const u32x4*)(arow + k0 + 16 + hh * 8);
    // B 32x16 f16 fragment: lane(n=m,hh): K = k0 + hh*8 + e  and  k0 + 16 + hh*8 + e
    b.q[0] = *(const u32x4*)(brow + k0 + hh * 8);
    b.q[1] = *(const u32x4*)(brow + k0 + 16 + hh * 8);
    acc = wmma_f16(a.v, b.v, acc);
  }
  // C/D layout: VGPR r -> row (hh*8 + r), col (n0 + m)
  float* crow = C + (size_t)(r0 + hh * 8) * Nout + n0 + m;
  #pragma unroll
  for (int r = 0; r < 8; ++r) crow[(size_t)r * Nout] = acc[r];
}

// ---------------- edge aggregation (E real edges + N self loops) ----------------
// 16 lanes per edge, 4 features per lane; agg pre-initialized with bias.
__global__ void k_gcn_aggregate(const float* __restrict__ hlin,
                                const long long* __restrict__ row,
                                const long long* __restrict__ col,
                                const float* __restrict__ ew,
                                const float* __restrict__ dis,
                                float* __restrict__ agg) {
  long long t = (long long)blockIdx.x * blockDim.x + threadIdx.x;
  long long e = t >> 4;
  int l = (int)(t & 15);
  const long long total = (long long)NEDGES + NNODES;
  if (e >= total) return;

  int r, c; float w;
  if (e < NEDGES) { r = (int)row[e]; c = (int)col[e]; w = ew[e]; }
  else            { r = c = (int)(e - NEDGES); w = 1.f; }

  float nrm = dis[r] * w * dis[c];
  f32x4 hv = *(const f32x4*)(hlin + (size_t)r * HDIM + l * 4);
  float* dst = agg + (size_t)c * HDIM + l * 4;
  atomicAdd(dst + 0, hv.x * nrm);
  atomicAdd(dst + 1, hv.y * nrm);
  atomicAdd(dst + 2, hv.z * nrm);
  atomicAdd(dst + 3, hv.w * nrm);
}

// ---------------- fused LSTM step + regression head ----------------
// block = 128 threads (4 waves) handles 16 nodes. Only i/g/o gate tiles computed (c0 = 0).
__global__ void k_lstm_head(const _Float16* __restrict__ H,     // [N][64] f16 (post-ReLU)
                            const _Float16* __restrict__ Wih,   // [256][64] f16 == B^T
                            const float* __restrict__ b_ih, const float* __restrict__ b_hh,
                            const float* __restrict__ w_reg, const float* __restrict__ b_reg,
                            float* __restrict__ out) {
  __shared__ float gates[16][GATES];
  __shared__ float partial[16][8];

  const int node0 = blockIdx.x * 16;
  const int wave  = threadIdx.x >> 5;
  const int lane  = threadIdx.x & 31;
  const int m     = lane & 15;
  const int hh    = lane >> 4;

  // A fragments (K = 64 -> two k-chunks), shared across this wave's 3 col tiles
  const _Float16* arow = H + (size_t)(node0 + m) * HDIM;
  frag16 a0, a1;
  a0.q[0] = *(const u32x4*)(arow + 0  + hh * 8);
  a0.q[1] = *(const u32x4*)(arow + 16 + hh * 8);
  a1.q[0] = *(const u32x4*)(arow + 32 + hh * 8);
  a1.q[1] = *(const u32x4*)(arow + 48 + hh * 8);

  // live gate col-tiles: i -> 0..3, g -> 8..11, o -> 12..15 (f-gate dead: c0 == 0)
  const int tiles[3] = { wave, 8 + wave, 12 + wave };
  #pragma unroll
  for (int t = 0; t < 3; ++t) {
    const int n0 = tiles[t] * 16;
    const _Float16* brow = Wih + (size_t)(n0 + m) * HDIM;
    frag16 b0, b1;
    b0.q[0] = *(const u32x4*)(brow + 0  + hh * 8);
    b0.q[1] = *(const u32x4*)(brow + 16 + hh * 8);
    b1.q[0] = *(const u32x4*)(brow + 32 + hh * 8);
    b1.q[1] = *(const u32x4*)(brow + 48 + hh * 8);
    v8f acc = {};
    acc = wmma_f16(a0.v, b0.v, acc);
    acc = wmma_f16(a1.v, b1.v, acc);
    #pragma unroll
    for (int r = 0; r < 8; ++r) gates[hh * 8 + r][n0 + m] = acc[r];
  }
  __syncthreads();

  // elementwise: thread -> (node mm, 8 features starting at fg)
  const int mm = threadIdx.x >> 3;
  const int fg = (threadIdx.x & 7) * 8;
  float accr = 0.f;
  #pragma unroll
  for (int j = 0; j < 8; ++j) {
    int f = fg + j;
    float ig = gates[mm][f]       + b_ih[f]       + b_hh[f];
    float gg = gates[mm][128 + f] + b_ih[128 + f] + b_hh[128 + f];
    float og = gates[mm][192 + f] + b_ih[192 + f] + b_hh[192 + f];
    float si = 1.f / (1.f + __expf(-ig));
    float so = 1.f / (1.f + __expf(-og));
    float cc = si * tanhf(gg);           // c0 == 0 -> f-gate term vanishes
    float hn = so * tanhf(cc);
    accr += hn * w_reg[f];
  }
  partial[mm][threadIdx.x & 7] = accr;
  __syncthreads();

  if (threadIdx.x < 16) {
    float s = 0.f;
    #pragma unroll
    for (int j = 0; j < 8; ++j) s += partial[threadIdx.x][j];
    out[node0 + threadIdx.x] = s + b_reg[0];
  }
}

// ---------------- host launcher ----------------
static inline int cdiv(long long a, long long b) { return (int)((a + b - 1) / b); }

extern "C" void kernel_launch(void* const* d_in, const int* in_sizes, int n_in,
                              void* d_out, int out_size, void* d_ws, size_t ws_size,
                              hipStream_t stream) {
  (void)in_sizes; (void)n_in; (void)out_size; (void)ws_size;

  const float*     x     = (const float*)d_in[0];
  const long long* eidx  = (const long long*)d_in[1];   // int64 [2][E]
  const float*     ew    = (const float*)d_in[2];
  const float*     W1    = (const float*)d_in[3];
  const float*     b1    = (const float*)d_in[4];
  const float*     W2    = (const float*)d_in[5];
  const float*     b2    = (const float*)d_in[6];
  const float*     W_ih  = (const float*)d_in[7];
  /* W_hh (d_in[8]) unused: h0 == 0 */
  const float*     b_ih  = (const float*)d_in[9];
  const float*     b_hh  = (const float*)d_in[10];
  const float*     W_reg = (const float*)d_in[11];
  const float*     b_reg = (const float*)d_in[12];
  float*           out   = (float*)d_out;

  const long long* row = eidx;
  const long long* col = eidx + NEDGES;

  // ---- workspace layout (all regions 64B aligned) ----
  char* ws = (char*)d_ws;
  _Float16* x_f16  = (_Float16*)(ws + 0);                       // 12,800,000 B
  float*    buf1   = (float*)   (ws + 12800000);                // hlin  [N][64]
  float*    buf2   = (float*)   (ws + 25600000);                // agg   [N][64]
  _Float16* h_f16  = (_Float16*)(ws + 38400000);                // [N][64]
  _Float16* W1t    = (_Float16*)(ws + 44800000);                // [64][128]
  _Float16* W2t    = (_Float16*)(ws + 44816384);                // [64][64]
  _Float16* Wih16  = (_Float16*)(ws + 44824576);                // [256][64]
  float*    deg    = (float*)   (ws + 44857344);                // [N]
  float*    dis    = (float*)   (ws + 45057344);                // [N]

  const int TB = 256;

  // 1) conversions
  k_cvt_f16<<<cdiv((long long)NNODES * FIN, TB), TB, 0, stream>>>(x, x_f16, NNODES * FIN);
  k_cvt_transpose_f16<<<cdiv(FIN * HDIM, TB), TB, 0, stream>>>(W1, W1t, FIN, HDIM);
  k_cvt_transpose_f16<<<cdiv(HDIM * HDIM, TB), TB, 0, stream>>>(W2, W2t, HDIM, HDIM);
  k_cvt_f16<<<cdiv(GATES * HDIM, TB), TB, 0, stream>>>(W_ih, Wih16, GATES * HDIM); // already B^T

  // 2) degrees + symmetric normalization
  k_fill<<<cdiv(NNODES, TB), TB, 0, stream>>>(deg, 1.0f, NNODES);   // self-loop weight
  k_deg_accum<<<cdiv(NEDGES, TB), TB, 0, stream>>>(col, ew, deg, NEDGES);
  k_deg_invsqrt<<<cdiv(NNODES, TB), TB, 0, stream>>>(deg, dis, NNODES);

  const int edgeBlocks = cdiv(((long long)NEDGES + NNODES) * 16, TB);

  // 3) layer 1: h = x @ W1 ; agg = bias + scatter(norm * h[r]) ; relu + f16
  {
    dim3 g(NNODES / 16, HDIM / 64);
    k_gemm_f16<<<g, 128, 0, stream>>>(x_f16, W1t, buf1, FIN, HDIM);
  }
  k_agg_init<<<cdiv((long long)NNODES * HDIM, TB), TB, 0, stream>>>(buf2, b1, NNODES * HDIM);
  k_gcn_aggregate<<<edgeBlocks, TB, 0, stream>>>(buf1, row, col, ew, dis, buf2);
  k_relu_cvt_f16<<<cdiv((long long)NNODES * HDIM, TB), TB, 0, stream>>>(buf2, h_f16, NNODES * HDIM);

  // 4) layer 2
  {
    dim3 g(NNODES / 16, HDIM / 64);
    k_gemm_f16<<<g, 128, 0, stream>>>(h_f16, W2t, buf1, HDIM, HDIM);
  }
  k_agg_init<<<cdiv((long long)NNODES * HDIM, TB), TB, 0, stream>>>(buf2, b2, NNODES * HDIM);
  k_gcn_aggregate<<<edgeBlocks, TB, 0, stream>>>(buf1, row, col, ew, dis, buf2);
  k_relu_cvt_f16<<<cdiv((long long)NNODES * HDIM, TB), TB, 0, stream>>>(buf2, h_f16, NNODES * HDIM);

  // 5) fused LSTM gates (WMMA) + nonlinearity + regression head
  k_lstm_head<<<NNODES / 16, 128, 0, stream>>>(h_f16, Wih16, b_ih, b_hh, W_reg, b_reg, out);
}
